// MontageAwareEmbedding_16638703305044
// MI455X (gfx1250) — compile-verified, matrix operand order checked by reading, not run on hardware
//
#include <hip/hip_runtime.h>
#include <hip/hip_bf16.h>
#include <math.h>
#include <stdint.h>

// ---------------------------------------------------------------------------
// MontageAwareEmbedding fused for gfx1250 (MI455X, wave32, WMMA f16->f32)
//   B=1024, L=64, C_IN=256, D=256, K=8
// Weight panels staged into LDS by the Tensor Data Mover (one descriptor per
// 64 KB K-half, TENSORcnt-tracked), double-buffered so the DMA of half h+1
// overlaps the WMMA stream of half h.  Falls back to
// global_load_async_to_lds_b128 (ASYNCcnt) if the TDM builtin is absent.
// ---------------------------------------------------------------------------
#define BB   1024
#define LL   64
#define CIN  256
#define DD   256
#define KK   8

#define ASTR 264      // activation LDS row stride (halves): 528 B, bank-staggered
#define WSTR 136      // weight LDS row stride (halves): 272 B, bank-staggered

typedef __attribute__((ext_vector_type(16))) _Float16 v16h;
typedef __attribute__((ext_vector_type(8)))  _Float16 v8h;
typedef __attribute__((ext_vector_type(8)))  float    v8f;
typedef __attribute__((ext_vector_type(4)))  unsigned v4u;
typedef __attribute__((ext_vector_type(8)))  int      v8i;
typedef __attribute__((ext_vector_type(4)))  int      v4i;

union FragH { v16h v; v8h h[2]; };

#if defined(__has_builtin)
#if __has_builtin(__builtin_amdgcn_tensor_load_to_lds)
#define HAVE_TDM 1
#endif
#endif
#ifndef HAVE_TDM
#define HAVE_TDM 0
#endif

__device__ __forceinline__ float gelu_f(float x) {
    return 0.5f * x * (1.0f + erff(x * 0.70710678118654752f));
}

// DMA one K-half of a weight panel (256 N-rows x 64 dwords = 64 KB) into LDS
// with a 4-dword pad after every row (matches WSTR).  Issued by wave 0 only.
__device__ __forceinline__ void tdm_load_half(
    const _Float16* __restrict__ g, int gstr_halves, unsigned lds_base, int lane)
{
#if HAVE_TDM
    (void)lane;
    uint64_t ga = (uint64_t)(uintptr_t)g;
    v4u g0;
    g0.x = 1u;                                            // count=1 (valid user D#)
    g0.y = lds_base;                                      // lds_addr
    g0.z = (unsigned)ga;                                  // global_addr[31:0]
    g0.w = ((unsigned)(ga >> 32) & 0x01FFFFFFu)           // global_addr[56:32]
         | (2u << 30);                                    // type = 2 ("image")
    v8i g1;
    g1[0] = (2 << 16)      // data_size = 4 bytes
          | (1 << 20)      // pad_enable
          | (5 << 22)      // pad_interval: 2^(5+1) = 64 dwords
          | (3 << 25);     // pad_amount: 3+1 = 4 dwords (16 B row pad)
    g1[1] = (int)(64u << 16);        // tensor_dim0 = 64 dwords (bits 79:48 lo)
    g1[2] = (int)(256u << 16);       // tensor_dim1 = 256 rows  (bits 111:80 lo)
    g1[3] = (int)(64u << 16);        // tile_dim0 = 64 dwords   (bits 127:112)
    g1[4] = 256;                     // tile_dim1 = 256 rows; tile_dim2 = 0
    g1[5] = gstr_halves >> 1;        // tensor_dim0_stride in dwords
    g1[6] = 0;
    g1[7] = 0;
    v4i z4 = {0, 0, 0, 0};
#if __clang_major__ >= 23
    v8i z8 = {0, 0, 0, 0, 0, 0, 0, 0};
    __builtin_amdgcn_tensor_load_to_lds(g0, g1, z4, z4, z8, 0);
#else
    __builtin_amdgcn_tensor_load_to_lds(g0, g1, z4, z4, 0);
#endif
#else
    // Fallback: wave-0 lanes issue async B128 copies (ASYNCcnt).
    for (int idx = lane; idx < 256 * 16; idx += 32) {
        int row = idx >> 4;
        int ch  = idx & 15;
        const _Float16* gp = g + (size_t)row * gstr_halves + ch * 8;
        unsigned lds = lds_base + (unsigned)(row * WSTR + ch * 8) * 2u;
        asm volatile("global_load_async_to_lds_b128 %0, %1, off"
                     :: "v"(lds), "v"(gp)
                     : "memory");
    }
#endif
}

__device__ __forceinline__ void tdm_wait0() {
#if HAVE_TDM
    __builtin_amdgcn_s_wait_tensorcnt(0);
#else
    asm volatile("s_wait_asynccnt 0x0" ::: "memory");
#endif
}

// One K-half (128 of K) of the wave tile: 4 k-steps, A and B both from LDS.
// All 8 B fragments are fetched before the 8 WMMAs so ds loads overlap.
__device__ __forceinline__ void compute_half(
    const _Float16* __restrict__ A, int astr, int akoff,
    const _Float16 (*wl)[WSTR],
    int mbase, int nbase, int r, int hi, v8f* acc)
{
#pragma unroll
    for (int kk = 0; kk < 4; ++kk) {
        const int k0 = kk * 32;
        FragH a;
        const _Float16* ap = A + (mbase + r) * astr + akoff + k0 + hi * 8;
        a.h[0] = *(const v8h*)(ap);
        a.h[1] = *(const v8h*)(ap + 16);
        FragH b[8];
#pragma unroll
        for (int t = 0; t < 8; ++t) {
            const _Float16* bp = &wl[nbase + t * 16 + r][k0 + hi * 8];
            b[t].h[0] = *(const v8h*)(bp);
            b[t].h[1] = *(const v8h*)(bp + 16);
        }
#pragma unroll
        for (int t = 0; t < 8; ++t)
            acc[t] = __builtin_amdgcn_wmma_f32_16x16x32_f16(
                false, a.v, false, b[t].v, (short)0, acc[t], false, false);
    }
}

struct HalfSrc { const _Float16* g; const _Float16* A; int akoff; };

// Software-pipelined GEMM: TDM-copy half h+1 while WMMA-ing half h.
__device__ __forceinline__ void gemm_pipelined(
    const HalfSrc* hs, int nh, int gstr, int astr,
    _Float16 (*w0)[WSTR], _Float16 (*w1)[WSTR],
    int wave, int lane, int mbase, int nbase, int r, int hi, v8f* acc)
{
    const unsigned lds_w0 = (unsigned)(uintptr_t)&w0[0][0];
    const unsigned lds_w1 = (unsigned)(uintptr_t)&w1[0][0];
    if (wave == 0) {
        tdm_load_half(hs[0].g, gstr, lds_w0, lane);
        tdm_wait0();
    }
    __syncthreads();
#pragma unroll
    for (int h = 0; h < 4; ++h) {
        if (h >= nh) break;
        _Float16 (*cur)[WSTR] = (h & 1) ? w1 : w0;
        unsigned  nxt_lds     = (h & 1) ? lds_w0 : lds_w1;
        if (h + 1 < nh && wave == 0)
            tdm_load_half(hs[h + 1].g, gstr, nxt_lds, lane);
        compute_half(hs[h].A, astr, hs[h].akoff, cur, mbase, nbase, r, hi, acc);
        if (wave == 0) tdm_wait0();
        __syncthreads();
    }
}

// ---------------------------------------------------------------------------
// Stage 0: one block, 256 threads.  All B-independent math + weight f16
// transposes.  thread t owns feature column d = t (D == 256).
// ---------------------------------------------------------------------------
__global__ __launch_bounds__(256) void precompute_kernel(
    const float* __restrict__ coords, const float* __restrict__ knn_d2,
    const float* __restrict__ knn_rel,
    const int* __restrict__ hemi, const int* __restrict__ ap,
    const int* __restrict__ si,
    const float* __restrict__ Wv, const float* __restrict__ bv,
    const float* __restrict__ W1, const float* __restrict__ b1,
    const float* __restrict__ W2, const float* __restrict__ b2,
    const float* __restrict__ coord_scale,
    const float* __restrict__ hemi_emb, const float* __restrict__ ap_emb,
    const float* __restrict__ si_emb,
    const float* __restrict__ region_scale,
    const float* __restrict__ Wf, const float* __restrict__ bf,
    const float* __restrict__ film_scale, const float* __restrict__ graph_sigma,
    const float* __restrict__ Wp1, const float* __restrict__ Wp2,
    const float* __restrict__ Wg,
    const float* __restrict__ We1, const float* __restrict__ be1,
    const float* __restrict__ We2, const float* __restrict__ be2,
    float* __restrict__ scale_g, float* __restrict__ rbias_g,
    float* __restrict__ eagg_g, float* __restrict__ pe_g,
    float* __restrict__ wgt_g, float* __restrict__ he_g,
    float* __restrict__ ch_g, float* __restrict__ ce_g,
    _Float16* __restrict__ WvT, _Float16* __restrict__ Wp1T,
    _Float16* __restrict__ Wp2T, _Float16* __restrict__ WgT)
{
    __shared__ float ff[LL][40];     // fourier features (39 used)
    __shared__ float wsh[LL][KK];    // normalized graph weights
    __shared__ float swsum[LL];      // sum_k w (for be2 fold)

    const int tid = threadIdx.x;

    if (tid < LL) {
        const int l = tid;
        const float twopi = 6.283185307179586f;
        for (int j = 0; j < 3; ++j) ff[l][j] = coords[l * 3 + j];
        for (int j = 0; j < 3; ++j) {
            float cv = ff[l][j];
            for (int f = 0; f < 6; ++f) {
                float ang = twopi * cv * (float)(1 << f);
                ff[l][3 + j * 6 + f]  = sinf(ang);
                ff[l][21 + j * 6 + f] = cosf(ang);
            }
        }
        float sg  = fmaxf(graph_sigma[0], 0.001f);
        float inv = 1.0f / (2.0f * sg * sg);
        float wk[KK]; float s = 0.0f;
        for (int k = 0; k < KK; ++k) { wk[k] = expf(-knn_d2[l*KK+k]*inv); s += wk[k]; }
        float dn = 1.0f / (s + 1e-6f);
        float ss = 0.0f;
        for (int k = 0; k < KK; ++k) {
            float w = wk[k] * dn;
            wsh[l][k] = w; wgt_g[l*KK+k] = w; ss += w;
        }
        swsum[l] = ss;
    }
    __syncthreads();

    const int d = tid;   // feature column

    // coord hidden = gelu(ff @ W1 + b1)
    for (int l = 0; l < LL; ++l) {
        float acc = b1[d];
        for (int j = 0; j < 39; ++j) acc += ff[l][j] * W1[j * DD + d];
        ch_g[l * DD + d] = gelu_f(acc);
    }
    __threadfence(); __syncthreads();

    // coord_emb = ch @ W2 + b2
    for (int l = 0; l < LL; ++l) {
        float acc = b2[d];
        for (int k = 0; k < DD; ++k) acc += ch_g[l * DD + k] * W2[k * DD + d];
        ce_g[l * DD + d] = acc;
    }
    __threadfence(); __syncthreads();

    // FiLM fold -> per-(l,d) affine on (x @ Wv)
    const float cs = coord_scale[0], rs = region_scale[0], fs = film_scale[0];
    for (int l = 0; l < LL; ++l) {
        float g = bf[d], bb = bf[DD + d];
        for (int k = 0; k < DD; ++k) {
            float cev = ce_g[l * DD + k];
            g  += cev * Wf[k * 2 * DD + d];
            bb += cev * Wf[k * 2 * DD + DD + d];
        }
        float sc  = 1.0f + fs * tanhf(g);
        float add = bv[d] + cs * ce_g[l * DD + d]
                  + rs * (hemi_emb[hemi[l]*DD + d] + ap_emb[ap[l]*DD + d]
                        + si_emb[si[l]*DD + d]);
        scale_g[l * DD + d] = sc;
        rbias_g[l * DD + d] = add * sc + fs * bb;
    }

    // edge MLP hidden: he[l*K+k][d] = gelu(rel @ We1 + be1)
    for (int rrow = 0; rrow < LL * KK; ++rrow) {
        const float* rel = knn_rel + rrow * 3;
        float acc = be1[d] + rel[0] * We1[0*DD + d] + rel[1] * We1[1*DD + d]
                           + rel[2] * We1[2*DD + d];
        he_g[rrow * DD + d] = gelu_f(acc);
    }
    __threadfence(); __syncthreads();

    // aggregate hidden with graph weights (reuse ch_g)
    for (int l = 0; l < LL; ++l) {
        float acc = 0.0f;
        for (int k = 0; k < KK; ++k)
            acc += wsh[l][k] * he_g[(l * KK + k) * DD + d];
        ch_g[l * DD + d] = acc;
    }
    __threadfence(); __syncthreads();

    // e_agg = eh @ We2 + be2 * sum_k w
    for (int l = 0; l < LL; ++l) {
        float acc = be2[d] * swsum[l];
        for (int m = 0; m < DD; ++m) acc += ch_g[l * DD + m] * We2[m * DD + d];
        eagg_g[l * DD + d] = acc;
    }

    // sinusoidal positional embedding
    {
        int i = d >> 1;
        float div = expf((float)(2 * i) * (-9.210340371976184f / (float)DD));
        for (int l = 0; l < LL; ++l) {
            float a = (float)l * div;
            pe_g[l * DD + d] = (d & 1) ? cosf(a) : sinf(a);
        }
    }

    // f16 weight transposes: WT[n][k] = (half)W[k][n]  (B-operand layout)
    for (int i = tid; i < DD * DD; i += 256) {
        int n = i >> 8, k = i & 255;
        WvT [n * DD + k] = (_Float16)Wv [k * DD + n];
        Wp1T[n * DD + k] = (_Float16)Wp1[k * DD + n];
        Wp2T[n * DD + k] = (_Float16)Wp2[k * DD + n];
    }
    for (int i = tid; i < DD * 2 * DD; i += 256) {
        int n = i >> 9, k = i & 511;
        WgT[n * 2 * DD + k] = (_Float16)Wg[k * DD + n];
    }
}

// ---------------------------------------------------------------------------
// Stage 1: one block per batch b.  All 4 GEMMs + gather fused.
// Activations + weight panels LDS-resident; weights DMA'd by the TDM.
// ---------------------------------------------------------------------------
__global__ __launch_bounds__(256) void fused_kernel(
    const float* __restrict__ x, const int* __restrict__ knn_idx,
    const float* __restrict__ scale_g, const float* __restrict__ rbias_g,
    const float* __restrict__ eagg_g, const float* __restrict__ pe_g,
    const float* __restrict__ wgt_g,
    const _Float16* __restrict__ WvT, const _Float16* __restrict__ Wp1T,
    const _Float16* __restrict__ Wp2T, const _Float16* __restrict__ WgT,
    const float* __restrict__ bp1, const float* __restrict__ bp2,
    const float* __restrict__ bg, const float* __restrict__ graph_scale,
    float* __restrict__ out)
{
    __shared__ _Float16 a0[LL][ASTR];    // nbh  -> nb2h
    __shared__ _Float16 a1[LL][ASTR];    // v1h  (post-FiLM activations)
    __shared__ _Float16 a2[LL][ASTR];    // xh   -> hh
    __shared__ _Float16 w0[DD][WSTR];    // weight panel ping
    __shared__ _Float16 w1[DD][WSTR];    // weight panel pong
    __shared__ int   sidx[LL * KK];
    __shared__ float sw  [LL * KK];

    const int tid  = threadIdx.x;
    const int b    = blockIdx.x;
    const int lane = tid & 31;
    const int wave = tid >> 5;
    const int r    = lane & 15;
    const int hi   = lane >> 4;
    const int mbase = (wave >> 1) * 16;   // 4 M-tiles
    const int nbase = (wave & 1) * 128;   // 2 N-halves

    // stage x[b] into LDS as f16
    const float* xb = x + (size_t)b * LL * CIN;
    for (int i = tid; i < LL * CIN; i += 256)
        a2[i >> 8][i & 255] = (_Float16)xb[i];
    for (int i = tid; i < LL * KK; i += 256) {
        sidx[i] = knn_idx[i];
        sw[i]   = wgt_g[i];
    }
    __syncthreads();

    v8f acc[8];

    // ---- GEMM1: v1 = affine(x @ Wv) -------------------------------------
#pragma unroll
    for (int t = 0; t < 8; ++t)
#pragma unroll
        for (int j = 0; j < 8; ++j) acc[t][j] = 0.0f;
    {
        HalfSrc hs[2] = { { WvT,       &a2[0][0], 0   },
                          { WvT + 128, &a2[0][0], 128 } };
        gemm_pipelined(hs, 2, DD, ASTR, w0, w1, wave, lane, mbase, nbase, r, hi, acc);
    }
#pragma unroll
    for (int t = 0; t < 8; ++t) {
        int n = nbase + t * 16 + r;
#pragma unroll
        for (int vr = 0; vr < 8; ++vr) {
            int m = mbase + vr + 8 * hi;
            float val = acc[t][vr] * scale_g[m * DD + n] + rbias_g[m * DD + n];
            a1[m][n] = (_Float16)val;
        }
    }
    __syncthreads();

    // ---- gather: nb = sum_k w * v1[idx] + e_agg -------------------------
    for (int i = tid; i < LL * DD; i += 256) {
        int l = i >> 8, dd = i & 255;
        float s = eagg_g[i];
#pragma unroll
        for (int k = 0; k < KK; ++k)
            s += sw[l * KK + k] * (float)a1[sidx[l * KK + k]][dd];
        a0[l][dd] = (_Float16)s;
    }
    __syncthreads();

    // ---- GEMM2: h = gelu(nb @ Wp1 + bp1) --------------------------------
#pragma unroll
    for (int t = 0; t < 8; ++t)
#pragma unroll
        for (int j = 0; j < 8; ++j) acc[t][j] = 0.0f;
    {
        HalfSrc hs[2] = { { Wp1T,       &a0[0][0], 0   },
                          { Wp1T + 128, &a0[0][0], 128 } };
        gemm_pipelined(hs, 2, DD, ASTR, w0, w1, wave, lane, mbase, nbase, r, hi, acc);
    }
#pragma unroll
    for (int t = 0; t < 8; ++t) {
        int n = nbase + t * 16 + r;
        float bn = bp1[n];
#pragma unroll
        for (int vr = 0; vr < 8; ++vr) {
            int m = mbase + vr + 8 * hi;
            a2[m][n] = (_Float16)gelu_f(acc[t][vr] + bn);
        }
    }
    __syncthreads();

    // ---- GEMM3: nb2 = h @ Wp2 + bp2 -------------------------------------
#pragma unroll
    for (int t = 0; t < 8; ++t)
#pragma unroll
        for (int j = 0; j < 8; ++j) acc[t][j] = 0.0f;
    {
        HalfSrc hs[2] = { { Wp2T,       &a2[0][0], 0   },
                          { Wp2T + 128, &a2[0][0], 128 } };
        gemm_pipelined(hs, 2, DD, ASTR, w0, w1, wave, lane, mbase, nbase, r, hi, acc);
    }
#pragma unroll
    for (int t = 0; t < 8; ++t) {
        int n = nbase + t * 16 + r;
        float bn = bp2[n];
#pragma unroll
        for (int vr = 0; vr < 8; ++vr) {
            int m = mbase + vr + 8 * hi;
            a0[m][n] = (_Float16)(acc[t][vr] + bn);
        }
    }
    __syncthreads();

    // ---- GEMM4: gate = sigmoid([v1 | nb2] @ Wg + bg); final combine -----
#pragma unroll
    for (int t = 0; t < 8; ++t)
#pragma unroll
        for (int j = 0; j < 8; ++j) acc[t][j] = 0.0f;
    {
        HalfSrc hs[4] = { { WgT,       &a1[0][0], 0   },
                          { WgT + 128, &a1[0][0], 128 },
                          { WgT + 256, &a0[0][0], 0   },
                          { WgT + 384, &a0[0][0], 128 } };
        gemm_pipelined(hs, 4, 2 * DD, ASTR, w0, w1, wave, lane, mbase, nbase, r, hi, acc);
    }
    const float gs = graph_scale[0];
#pragma unroll
    for (int t = 0; t < 8; ++t) {
        int n = nbase + t * 16 + r;
        float bn = bg[n];
#pragma unroll
        for (int vr = 0; vr < 8; ++vr) {
            int m = mbase + vr + 8 * hi;
            float gate = 1.0f / (1.0f + expf(-(acc[t][vr] + bn)));
            float v    = (float)a1[m][n];
            float nb2  = (float)a0[m][n];
            out[((size_t)b * LL + m) * DD + n] =
                v + gs * gate * nb2 + pe_g[m * DD + n];
        }
    }
}

// ---------------------------------------------------------------------------
extern "C" void kernel_launch(void* const* d_in, const int* in_sizes, int n_in,
                              void* d_out, int out_size, void* d_ws, size_t ws_size,
                              hipStream_t stream) {
    const float* x           = (const float*)d_in[0];
    const float* coords      = (const float*)d_in[1];
    const float* knn_d2      = (const float*)d_in[2];
    const float* knn_rel     = (const float*)d_in[3];
    const int*   knn_idx     = (const int*)  d_in[4];
    const int*   hemi_id     = (const int*)  d_in[5];
    const int*   ap_id       = (const int*)  d_in[6];
    const int*   si_id       = (const int*)  d_in[7];
    const float* Wv          = (const float*)d_in[8];
    const float* bv          = (const float*)d_in[9];
    const float* W1          = (const float*)d_in[10];
    const float* b1          = (const float*)d_in[11];
    const float* W2          = (const float*)d_in[12];
    const float* b2          = (const float*)d_in[13];
    const float* coord_scale = (const float*)d_in[14];
    const float* hemi_emb    = (const float*)d_in[15];
    const float* ap_emb      = (const float*)d_in[16];
    const float* si_emb      = (const float*)d_in[17];
    const float* region_scale= (const float*)d_in[18];
    const float* Wf          = (const float*)d_in[19];
    const float* bf          = (const float*)d_in[20];
    const float* film_scale  = (const float*)d_in[21];
    const float* graph_sigma = (const float*)d_in[22];
    const float* graph_scale = (const float*)d_in[23];
    const float* Wp1         = (const float*)d_in[24];
    const float* bp1         = (const float*)d_in[25];
    const float* Wp2         = (const float*)d_in[26];
    const float* bp2         = (const float*)d_in[27];
    const float* Wg          = (const float*)d_in[28];
    const float* bg          = (const float*)d_in[29];
    const float* We1         = (const float*)d_in[30];
    const float* be1         = (const float*)d_in[31];
    const float* We2         = (const float*)d_in[32];
    const float* be2         = (const float*)d_in[33];

    char* ws = (char*)d_ws;
    size_t o = 0;
    float* scale_g = (float*)(ws + o); o += LL * DD * 4;
    float* rbias_g = (float*)(ws + o); o += LL * DD * 4;
    float* eagg_g  = (float*)(ws + o); o += LL * DD * 4;
    float* pe_g    = (float*)(ws + o); o += LL * DD * 4;
    float* wgt_g   = (float*)(ws + o); o += LL * KK * 4;
    float* he_g    = (float*)(ws + o); o += LL * KK * DD * 4;
    float* ch_g    = (float*)(ws + o); o += LL * DD * 4;
    float* ce_g    = (float*)(ws + o); o += LL * DD * 4;
    _Float16* WvT  = (_Float16*)(ws + o); o += DD * DD * 2;
    _Float16* Wp1T = (_Float16*)(ws + o); o += DD * DD * 2;
    _Float16* Wp2T = (_Float16*)(ws + o); o += DD * DD * 2;
    _Float16* WgT  = (_Float16*)(ws + o); o += DD * 2 * DD * 2;

    precompute_kernel<<<1, 256, 0, stream>>>(
        coords, knn_d2, knn_rel, hemi_id, ap_id, si_id,
        Wv, bv, W1, b1, W2, b2, coord_scale,
        hemi_emb, ap_emb, si_emb, region_scale,
        Wf, bf, film_scale, graph_sigma,
        Wp1, Wp2, Wg, We1, be1, We2, be2,
        scale_g, rbias_g, eagg_g, pe_g, wgt_g, he_g, ch_g, ce_g,
        WvT, Wp1T, Wp2T, WgT);

    fused_kernel<<<BB, 256, 0, stream>>>(
        x, knn_idx, scale_g, rbias_g, eagg_g, pe_g, wgt_g,
        WvT, Wp1T, Wp2T, WgT, bp1, bp2, bg, graph_scale,
        (float*)d_out);
}
